// MambaEncoderBlock_66305705115705
// MI455X (gfx1250) — matrix operand order repaired and checked by reading, not compiled, not run on hardware
//
#include <hip/hip_runtime.h>
#include <hip/hip_bf16.h>

// ---------------------------------------------------------------------------
// Mamba encoder block for MI455X (gfx1250, wave32, WMMA + TDM).
// GEMMs run on v_wmma_f32_16x16x32_bf16 (bf16 in, f32 accumulate); the
// A-tile is staged into LDS by the Tensor Data Mover when available.
// ---------------------------------------------------------------------------

constexpr int kDModel = 512;
constexpr int kDInner = 1024;   // 2*D_MODEL
constexpr int kDState = 16;
constexpr int kDConv  = 4;
constexpr int kDtRank = 32;
constexpr int kB      = 4;
constexpr int kT      = 2048;
constexpr int kBT     = kB * kT;                // 8192 rows
constexpr int kDbl    = kDtRank + 2 * kDState;  // 64

typedef __attribute__((ext_vector_type(16))) __bf16        v16bf;
typedef __attribute__((ext_vector_type(8)))  float         v8f;
typedef __attribute__((ext_vector_type(4)))  unsigned int  uint4v;
typedef __attribute__((ext_vector_type(4)))  unsigned int  u32x4;
typedef __attribute__((ext_vector_type(8)))  int           i32x8;
typedef __attribute__((ext_vector_type(4)))  int           i32x4;

#if __has_builtin(__builtin_amdgcn_tensor_load_to_lds)
#define USE_TDM 1
#else
#define USE_TDM 0
#endif

union FragBF {
  v16bf  v;
  uint4v q[2];
};

static __device__ __forceinline__ unsigned short f32_to_bf16_rne(float f) {
  unsigned int u = __float_as_uint(f);
  unsigned int lsb = (u >> 16) & 1u;
  u += 0x7fffu + lsb;                 // round-to-nearest-even
  return (unsigned short)(u >> 16);
}

// ---------------------------------------------------------------------------
// fp32 -> bf16 conversion
// ---------------------------------------------------------------------------
__global__ void f32_to_bf16_kernel(const float* __restrict__ src,
                                   unsigned short* __restrict__ dst, int n) {
  int i = blockIdx.x * blockDim.x + threadIdx.x;
  if (i < n) dst[i] = f32_to_bf16_rne(src[i]);
}

// ---------------------------------------------------------------------------
// Generic bf16 GEMM: C[M,N] = A[M,K] * B[K,N], row-major, fp32 out.
// Block: 128 threads (4 waves). Block tile 64x64, wave tile 16x64, K-step 32.
// Requires M%64==0, N%64==0, K%32==0.
//
// Per-lane fragment layouts (CDNA5 ISA 7.12.2, 16-bit):
//   A (16x32): lane l (m = l&15): elems 0..7  = K[(l>=16?8:0) + 0..7]
//                                 elems 8..15 = K[(l>=16?8:0) + 16..23]
//   B (32x16): lane l (n = l&15): elems 0..7  = K[(l>=16?8:0) + 0..7]
//                                 elems 8..15 = K[(l>=16?8:0) + 16..23]
//   C/D (16x16 f32): vgpr r: row = (l>=16?8:0)+r, col = l&15
//
// A tile staged via TENSOR_LOAD_TO_LDS (TDM, wave 0 issues; TENSORcnt is
// per-wave so only wave 0 waits, then all meet at the workgroup barrier).
// ---------------------------------------------------------------------------
__global__ __launch_bounds__(128)
void gemm_bf16_wmma_kernel(const unsigned short* __restrict__ A,
                           const unsigned short* __restrict__ B,
                           float* __restrict__ C,
                           int M, int N, int K) {
  constexpr int TM = 64, TN = 64, TK = 32;
  __shared__ unsigned short As[TM * TK];    // [m][k], row stride 32
  __shared__ unsigned short BsT[TN * TK];   // [n][k], row stride 32 (transposed)

  const int tid   = threadIdx.x;
  const int lane  = tid & 31;
  const int wave  = tid >> 5;
  const int lhalf = (lane >> 4) & 1;        // 0: lanes 0-15, 1: lanes 16-31
  const int lmod  = lane & 15;

  const int mBlock = blockIdx.y * TM;
  const int nBlock = blockIdx.x * TN;
  const int mWave  = mBlock + wave * 16;

  v8f zero = {0.f, 0.f, 0.f, 0.f, 0.f, 0.f, 0.f, 0.f};
  v8f acc[4];
#pragma unroll
  for (int i = 0; i < 4; ++i) acc[i] = zero;

  const int aRow  = tid >> 1;   // 0..63
  const int aHalf = tid & 1;    // 16 bf16 each
  const int bRow  = tid >> 2;   // 0..31 (k)
  const int bQuad = tid & 3;    // 16 n-columns each

#if USE_TDM
  // ---- Tensor DMA descriptor for the 64x32 bf16 A tile (2D, row stride K).
  //      group0: count=1 | lds_addr | global_addr(57b) | type=2
  //      group1: data_size=2B, tensor_dim0=32, tensor_dim1=64,
  //              tile_dim0=32, tile_dim1=64, tensor_dim0_stride=K
  const unsigned int ldsA = (unsigned int)(uintptr_t)(void*)As;  // LDS byte addr
  u32x4 g0 = {0u, 0u, 0u, 0u};
  i32x8 g1 = {0, 0, 0, 0, 0, 0, 0, 0};
  i32x4 gz4 = {0, 0, 0, 0};
  i32x8 gz8 = {0, 0, 0, 0, 0, 0, 0, 0};
  g0[0] = 1u;                       // count = 1 (valid user descriptor)
  g0[1] = ldsA;                     // lds_addr, bits 63:32
  g1[0] = 0x00010000;               // data_size = 1 -> 2 bytes
  g1[1] = (TK & 0xFFFF) << 16;      // tensor_dim0[15:0] in bits 63:48
  g1[2] = (TK >> 16) | ((TM & 0xFFFF) << 16);  // dim0 hi | tensor_dim1 lo
  g1[3] = (TM >> 16) | (TK << 16);  // dim1 hi | tile_dim0 = 32
  g1[4] = TM;                       // tile_dim1 = 64, tile_dim2 = 0
  g1[5] = K;                        // tensor_dim0_stride[31:0] = K elements
#endif

  for (int k0 = 0; k0 < K; k0 += TK) {
    // ---- stage A tile
#if USE_TDM
    if (wave == 0) {
      unsigned long long ga = (unsigned long long)(uintptr_t)A +
                              ((unsigned long long)mBlock * K + k0) * 2ull;
      g0[2] = (unsigned int)ga;                         // addr[31:0]
      g0[3] = (unsigned int)(ga >> 32) | 0x80000000u;   // addr[56:32] | type=2
      __builtin_amdgcn_tensor_load_to_lds(g0, g1, gz4, gz4, gz8, 0);
    }
#else
    {
      const uint4v* src = reinterpret_cast<const uint4v*>(
          A + (size_t)(mBlock + aRow) * K + k0 + aHalf * 16);
      uint4v* dst = reinterpret_cast<uint4v*>(&As[aRow * TK + aHalf * 16]);
      dst[0] = src[0];
      dst[1] = src[1];
    }
#endif
    // ---- stage B tile transposed: thread reads 16 contiguous n for fixed k
    {
      const unsigned short* src =
          B + (size_t)(k0 + bRow) * N + nBlock + bQuad * 16;
#pragma unroll
      for (int j = 0; j < 16; ++j)
        BsT[(bQuad * 16 + j) * TK + bRow] = src[j];
    }
    // prefetch next K tile while this one is consumed
    if (k0 + TK < K) {
#if !USE_TDM
      __builtin_prefetch(A + (size_t)(mBlock + aRow) * K + k0 + TK, 0, 1);
#endif
      __builtin_prefetch(B + (size_t)(k0 + TK + bRow) * N + nBlock, 0, 1);
    }
#if USE_TDM
    if (wave == 0) __builtin_amdgcn_s_wait_tensorcnt(0);
#endif
    __syncthreads();

    // ---- A fragment: two contiguous uint4 LDS loads
    FragBF fa;
    {
      const unsigned short* ap = &As[lmod * TK + lhalf * 8];
      fa.q[0] = *reinterpret_cast<const uint4v*>(ap);       // K base..base+7
      fa.q[1] = *reinterpret_cast<const uint4v*>(ap + 16);  // K base+16..+23
    }
    // ---- all 4 B fragments first, then 4 back-to-back WMMAs
    FragBF fb[4];
#pragma unroll
    for (int nt = 0; nt < 4; ++nt) {
      const unsigned short* bp = &BsT[(nt * 16 + lmod) * TK + lhalf * 8];
      fb[nt].q[0] = *reinterpret_cast<const uint4v*>(bp);
      fb[nt].q[1] = *reinterpret_cast<const uint4v*>(bp + 16);
    }
#pragma unroll
    for (int nt = 0; nt < 4; ++nt)
      acc[nt] = __builtin_amdgcn_wmma_f32_16x16x32_bf16(
          false, fa.v, false, fb[nt].v, (short)0, acc[nt], false, false);
    __syncthreads();
  }

  // ---- epilogue: C/D layout -> global
  const int rowOff = lhalf * 8;
#pragma unroll
  for (int nt = 0; nt < 4; ++nt) {
    const int col = nBlock + nt * 16 + lmod;
#pragma unroll
    for (int r = 0; r < 8; ++r)
      C[(size_t)(mWave + rowOff + r) * N + col] = acc[nt][r];
  }
}

// ---------------------------------------------------------------------------
// Causal depthwise conv (D_CONV=4) + SiLU. Reads xin = xz[:, :, 0:1024].
// Writes xc (fp32, for scan) and xcb (bf16, for GEMM with Wx).
// ---------------------------------------------------------------------------
__global__ void conv_silu_kernel(const float* __restrict__ xz,
                                 const float* __restrict__ conv_w,
                                 const float* __restrict__ conv_b,
                                 float* __restrict__ xc,
                                 unsigned short* __restrict__ xcb) {
  int idx = blockIdx.x * blockDim.x + threadIdx.x;   // over B*T*D_INNER
  if (idx >= kBT * kDInner) return;
  const int d  = idx & (kDInner - 1);
  const int bt = idx >> 10;                // b*T + t  (D_INNER = 2^10)
  const int t  = bt & (kT - 1);
  const int b  = bt >> 11;                 // T = 2^11

  float acc = conv_b[d];
#pragma unroll
  for (int k = 0; k < kDConv; ++k) {
    int ts = t + k - (kDConv - 1);
    if (ts >= 0)
      acc += xz[(size_t)(b * kT + ts) * (2 * kDInner) + d] * conv_w[d * kDConv + k];
  }
  float s = acc / (1.0f + __expf(-acc));   // SiLU
  xc[idx]  = s;
  xcb[idx] = f32_to_bf16_rne(s);
}

// ---------------------------------------------------------------------------
// Extract dt_low (first 32 cols of dbl) as compact bf16 (8192 x 32).
// ---------------------------------------------------------------------------
__global__ void extract_dtlow_kernel(const float* __restrict__ dbl,
                                     unsigned short* __restrict__ dst) {
  int i = blockIdx.x * blockDim.x + threadIdx.x;     // over kBT*32
  if (i >= kBT * kDtRank) return;
  int r = i >> 5, c = i & 31;
  dst[i] = f32_to_bf16_rne(dbl[(size_t)r * kDbl + c]);
}

// ---------------------------------------------------------------------------
// Selective scan. One thread per (b, d) channel; h[16] in registers.
// B_t/C_t (cols 32..63 of dbl) staged in LDS per 64-step chunk, shared by
// all channels of the batch. Fuses softplus, D-skip, SiLU(z) gate; emits
// bf16 y for the output GEMM.
// ---------------------------------------------------------------------------
__global__ __launch_bounds__(256)
void scan_kernel(const float* __restrict__ draw,   // (BT, 1024) pre-softplus
                 const float* __restrict__ dbl,    // (BT, 64)
                 const float* __restrict__ xc,     // (BT, 1024)
                 const float* __restrict__ xz,     // (BT, 2048), z at cols 1024+
                 const float* __restrict__ bdt,
                 const float* __restrict__ A_log,
                 const float* __restrict__ Dp,
                 unsigned short* __restrict__ yb) {
  __shared__ float bc[64][32];               // [tt][n]: 0..15 = B_t, 16..31 = C_t
  const int tid = threadIdx.x;
  const int d   = blockIdx.x * 256 + tid;
  const int b   = blockIdx.y;

  float A_d[kDState], h[kDState];
#pragma unroll
  for (int n = 0; n < kDState; ++n) {
    A_d[n] = -__expf(A_log[d * kDState + n]);
    h[n]   = 0.f;
  }
  const float bdt_d = bdt[d];
  const float Dp_d  = Dp[d];

  for (int t0 = 0; t0 < kT; t0 += 64) {
    for (int i = tid; i < 64 * 32; i += 256) {
      int tt = i >> 5, j = i & 31;
      bc[tt][j] = dbl[(size_t)(b * kT + t0 + tt) * kDbl + kDtRank + j];
    }
    __syncthreads();

    for (int tt = 0; tt < 64; ++tt) {
      const size_t row = (size_t)(b * kT) + t0 + tt;
      float dr    = draw[row * kDInner + d] + bdt_d;
      float delta = (dr > 20.f) ? dr : log1pf(__expf(dr));   // softplus
      float xv    = xc[row * kDInner + d];
      float dx    = delta * xv;
      float y     = 0.f;
#pragma unroll
      for (int n = 0; n < kDState; ++n) {
        float dA = __expf(delta * A_d[n]);
        h[n] = dA * h[n] + dx * bc[tt][n];
        y += h[n] * bc[tt][kDState + n];
      }
      y += xv * Dp_d;
      float z = xz[row * (2 * kDInner) + kDInner + d];
      y *= z / (1.f + __expf(-z));                            // * SiLU(z)
      yb[row * kDInner + d] = f32_to_bf16_rne(y);
    }
    __syncthreads();
  }
}

// ---------------------------------------------------------------------------
// Residual add + LayerNorm(512) + DS=2 downsample.
// One block per (b, t2) output row; 256 threads, 2 columns each, 2 LN rows.
// ---------------------------------------------------------------------------
__global__ __launch_bounds__(256)
void ln_ds_kernel(const float* __restrict__ mm,     // (BT, 512) y @ Wout
                  const float* __restrict__ xres,   // (BT, 512) residual
                  const float* __restrict__ g,
                  const float* __restrict__ bb,
                  const float* __restrict__ dsw,    // (2,)
                  const float* __restrict__ dsb,    // (1,)
                  float* __restrict__ out) {        // (B, T/2, 512)
  __shared__ float s1[256], s2[256];
  const int tid = threadIdx.x;
  const int blk = blockIdx.x;                // b*(T/2) + t2
  const int b   = blk >> 10;
  const int t2  = blk & 1023;
  const int c0  = tid, c1 = tid + 256;

  float hn[2][2];
  for (int r = 0; r < 2; ++r) {
    const size_t row = (size_t)b * kT + 2 * t2 + r;
    float v0 = mm[row * kDModel + c0] + xres[row * kDModel + c0];
    float v1 = mm[row * kDModel + c1] + xres[row * kDModel + c1];
    s1[tid] = v0 + v1;
    s2[tid] = v0 * v0 + v1 * v1;
    __syncthreads();
    for (int s = 128; s > 0; s >>= 1) {
      if (tid < s) { s1[tid] += s1[tid + s]; s2[tid] += s2[tid + s]; }
      __syncthreads();
    }
    float mu  = s1[0] * (1.0f / kDModel);
    float var = s2[0] * (1.0f / kDModel) - mu * mu;
    float rs  = rsqrtf(var + 1e-5f);
    hn[r][0] = (v0 - mu) * rs * g[c0] + bb[c0];
    hn[r][1] = (v1 - mu) * rs * g[c1] + bb[c1];
    __syncthreads();
  }
  const float w0 = dsw[0], w1 = dsw[1], db = dsb[0];
  const size_t orow = (size_t)blk * kDModel;
  out[orow + c0] = hn[0][0] * w0 + hn[1][0] * w1 + db;
  out[orow + c1] = hn[0][1] * w0 + hn[1][1] * w1 + db;
}

// ---------------------------------------------------------------------------
// Host launcher
// ---------------------------------------------------------------------------
extern "C" void kernel_launch(void* const* d_in, const int* in_sizes, int n_in,
                              void* d_out, int out_size, void* d_ws, size_t ws_size,
                              hipStream_t stream) {
  const float* x      = (const float*)d_in[0];
  const float* Win    = (const float*)d_in[1];
  const float* conv_w = (const float*)d_in[2];
  const float* conv_b = (const float*)d_in[3];
  const float* Wx     = (const float*)d_in[4];
  const float* Wdt    = (const float*)d_in[5];
  const float* bdt    = (const float*)d_in[6];
  const float* A_log  = (const float*)d_in[7];
  const float* Dp     = (const float*)d_in[8];
  const float* Wout   = (const float*)d_in[9];
  const float* ln_g   = (const float*)d_in[10];
  const float* ln_b   = (const float*)d_in[11];
  const float* ds_w   = (const float*)d_in[12];
  const float* ds_b   = (const float*)d_in[13];
  float* out = (float*)d_out;

  // workspace layout (256B aligned slices), ~199 MB total
  char* p = (char*)d_ws;
  auto alloc = [&](size_t bytes) {
    char* r = p;
    p += (bytes + 255) & ~(size_t)255;
    return r;
  };
  unsigned short* xb     = (unsigned short*)alloc((size_t)kBT * kDModel * 2);
  unsigned short* Winb   = (unsigned short*)alloc((size_t)kDModel * 2 * kDInner * 2);
  unsigned short* Wxb    = (unsigned short*)alloc((size_t)kDInner * kDbl * 2);
  unsigned short* Wdtb   = (unsigned short*)alloc((size_t)kDtRank * kDInner * 2);
  unsigned short* Woutb  = (unsigned short*)alloc((size_t)kDInner * kDModel * 2);
  float*          xzf    = (float*)alloc((size_t)kBT * 2 * kDInner * 4);
  float*          xcf    = (float*)alloc((size_t)kBT * kDInner * 4);
  unsigned short* xcb    = (unsigned short*)alloc((size_t)kBT * kDInner * 2);
  float*          dblf   = (float*)alloc((size_t)kBT * kDbl * 4);
  unsigned short* dtlowb = (unsigned short*)alloc((size_t)kBT * kDtRank * 2);
  float*          drawf  = (float*)alloc((size_t)kBT * kDInner * 4);
  unsigned short* ybb    = (unsigned short*)alloc((size_t)kBT * kDInner * 2);
  float*          mmf    = (float*)alloc((size_t)kBT * kDModel * 4);

  auto cvt = [&](const float* s, unsigned short* d, int n) {
    f32_to_bf16_kernel<<<(n + 255) / 256, 256, 0, stream>>>(s, d, n);
  };

  // 1) bf16 conversions of GEMM operands
  cvt(x,    xb,    kBT * kDModel);
  cvt(Win,  Winb,  kDModel * 2 * kDInner);
  cvt(Wx,   Wxb,   kDInner * kDbl);
  cvt(Wdt,  Wdtb,  kDtRank * kDInner);
  cvt(Wout, Woutb, kDInner * kDModel);

  // 2) xz = x @ Win : (8192 x 512) * (512 x 2048)
  gemm_bf16_wmma_kernel<<<dim3(2 * kDInner / 64, kBT / 64), 128, 0, stream>>>(
      xb, Winb, xzf, kBT, 2 * kDInner, kDModel);

  // 3) causal conv + SiLU -> xc
  conv_silu_kernel<<<(kBT * kDInner + 255) / 256, 256, 0, stream>>>(
      xzf, conv_w, conv_b, xcf, xcb);

  // 4) dbl = xc @ Wx : (8192 x 1024) * (1024 x 64)
  gemm_bf16_wmma_kernel<<<dim3(kDbl / 64, kBT / 64), 128, 0, stream>>>(
      xcb, Wxb, dblf, kBT, kDbl, kDInner);

  // 5) dt_low -> compact bf16
  extract_dtlow_kernel<<<(kBT * kDtRank + 255) / 256, 256, 0, stream>>>(dblf, dtlowb);

  // 6) draw = dt_low @ Wdt : (8192 x 32) * (32 x 1024)
  gemm_bf16_wmma_kernel<<<dim3(kDInner / 64, kBT / 64), 128, 0, stream>>>(
      dtlowb, Wdtb, drawf, kBT, kDInner, kDtRank);

  // 7) selective scan + gates -> y (bf16)
  scan_kernel<<<dim3(kDInner / 256, kB), 256, 0, stream>>>(
      drawf, dblf, xcf, xzf, bdt, A_log, Dp, ybb);

  // 8) mm = y @ Wout : (8192 x 1024) * (1024 x 512)
  gemm_bf16_wmma_kernel<<<dim3(kDModel / 64, kBT / 64), 128, 0, stream>>>(
      ybb, Woutb, mmf, kBT, kDModel, kDInner);

  // 9) residual + LayerNorm + DS=2 downsample
  ln_ds_kernel<<<kB * (kT / 2), 256, 0, stream>>>(
      mmf, x, ln_g, ln_b, ds_w, ds_b, out);

  (void)in_sizes; (void)n_in; (void)out_size; (void)ws_size;
}
